// AttentionDisaggregated_32633161515364
// MI455X (gfx1250) — matrix operand order repaired and checked by reading, not compiled, not run on hardware
//
#include <hip/hip_runtime.h>
#include <hip/hip_bf16.h>

// ---------------------------------------------------------------------------
// Problem constants (from reference)
// ---------------------------------------------------------------------------
#define DM 2048   // d_model
#define NH 16     // n_heads
#define HD 128    // head_dim
#define BB 2      // batch
#define SS 2048   // seq len
#define MTOT (BB*SS)          // 4096 rows
#define PRE_OUT ((size_t)BB*SS*DM)  // 8388608 floats of prefill output

typedef __attribute__((ext_vector_type(8)))  __bf16 v8bf;
typedef __attribute__((ext_vector_type(16))) __bf16 v16bf;
typedef __attribute__((ext_vector_type(8)))  float  v8f;

#define DEV __device__ __forceinline__

DEV v16bf cat16(v8bf a, v8bf b) {
  v16bf r;
#pragma unroll
  for (int i = 0; i < 8; i++) { r[i] = a[i]; r[i + 8] = b[i]; }
  return r;
}

DEV v8f wmma_bf16(v16bf a, v16bf b, v8f c) {
  // D(16x16 f32) = A(16x32 bf16) x B(32x16 bf16) + C
  return __builtin_amdgcn_wmma_f32_16x16x32_bf16(
      /*neg_a=*/false, a, /*neg_b=*/false, b,
      /*c_mod=*/(short)0, c, /*reuse_a=*/false, /*reuse_b=*/false);
}

// ---------------------------------------------------------------------------
// Elementwise f32 -> bf16
// ---------------------------------------------------------------------------
__global__ void k_cvt_bf16(const float* __restrict__ in, __bf16* __restrict__ out, int n) {
  int i = blockIdx.x * 256 + threadIdx.x;
  if (i < n) out[i] = (__bf16)in[i];
}

// ---------------------------------------------------------------------------
// Tiled transpose: out[n][d] = (bf16) in[d][n], square D x D
// ---------------------------------------------------------------------------
__global__ __launch_bounds__(256) void k_transpose_bf16(
    const float* __restrict__ in, __bf16* __restrict__ out, int D) {
  __shared__ float tile[32][33];
  int bx = blockIdx.x * 32, by = blockIdx.y * 32;
  int tx = threadIdx.x & 31;
  int ty = threadIdx.x >> 5;  // 0..7
  for (int j = ty; j < 32; j += 8)
    tile[j][tx] = in[(size_t)(by + j) * D + bx + tx];
  __syncthreads();
  for (int j = ty; j < 32; j += 8)
    out[(size_t)(bx + j) * D + by + tx] = (__bf16)tile[tx][j];
}

// ---------------------------------------------------------------------------
// Generic bf16 WMMA GEMM:  C[M,N] = A[M,K] * Bt[N,K]^T   (Bt is n-major, k contig)
//   block: 256 threads = 8 waves (2 x 4), block tile 64(M) x 256(N)
//   wave tile: 32 x 64 (2x4 fragments, 8 WMMA / K-step of 32)
// Epilogue modes:
//   0: f32 row-major into dst[M,N]                (output projection)
//   1: bf16 into per-head layout [b,head][s][h]   (Q, K)
//   2: bf16 into transposed head layout [b,head][h][s]   (V)
// ---------------------------------------------------------------------------
__global__ __launch_bounds__(256) void k_gemm_bf16(
    const __bf16* __restrict__ A, const __bf16* __restrict__ Bt,
    void* __restrict__ dst, int M, int N, int K, int mode) {
  int lane = threadIdx.x & 31;
  int wave = threadIdx.x >> 5;
  int li = lane & 15, hi = lane >> 4;
  int wm = wave >> 2, wn = wave & 3;
  int m0 = blockIdx.y * 64 + wm * 32;
  int n0 = blockIdx.x * 256 + wn * 64;

  const __bf16* arow[2];
  const __bf16* brow[4];
#pragma unroll
  for (int i = 0; i < 2; i++)
    arow[i] = A + (size_t)(m0 + i * 16 + li) * K;
#pragma unroll
  for (int i = 0; i < 4; i++)
    brow[i] = Bt + (size_t)(n0 + i * 16 + li) * K;

  v8f acc[2][4];
#pragma unroll
  for (int i = 0; i < 2; i++)
#pragma unroll
    for (int j = 0; j < 4; j++)
#pragma unroll
      for (int r = 0; r < 8; r++) acc[i][j][r] = 0.0f;

  for (int kc = 0; kc < K; kc += 32) {
    v16bf af[2], bfr[4];
#pragma unroll
    for (int i = 0; i < 2; i++) {
      // A fragment (16x32): lane<16 holds K {0..7,16..23}, lane>=16 {8..15,24..31}
      const __bf16* p = arow[i] + kc + hi * 8;
      af[i] = cat16(*(const v8bf*)p, *(const v8bf*)(p + 16));
    }
#pragma unroll
    for (int i = 0; i < 4; i++) {
      // B fragment (32x16): lane<16 -> N=li, K {0..7,16..23}; lane>=16 -> K {8..15,24..31}
      const __bf16* q = brow[i] + kc + hi * 8;
      bfr[i] = cat16(*(const v8bf*)q, *(const v8bf*)(q + 16));
    }
    if (kc + 128 < K) {
#pragma unroll
      for (int i = 0; i < 2; i++) __builtin_prefetch(arow[i] + kc + 128, 0, 1);
#pragma unroll
      for (int i = 0; i < 4; i++) __builtin_prefetch(brow[i] + kc + 128, 0, 1);
    }
#pragma unroll
    for (int mi = 0; mi < 2; mi++)
#pragma unroll
      for (int ni = 0; ni < 4; ni++)
        acc[mi][ni] = wmma_bf16(af[mi], bfr[ni], acc[mi][ni]);
  }

  // C layout: component r, lanes 0-15 -> M=r, lanes 16-31 -> M=r+8; N = li
#pragma unroll
  for (int mi = 0; mi < 2; mi++)
#pragma unroll
    for (int ni = 0; ni < 4; ni++)
#pragma unroll
      for (int r = 0; r < 8; r++) {
        int m = m0 + mi * 16 + r + hi * 8;
        int n = n0 + ni * 16 + li;
        float v = acc[mi][ni][r];
        if (mode == 0) {
          ((float*)dst)[(size_t)m * N + n] = v;
        } else {
          int b = m >> 11, s = m & 2047;   // m = b*SS + s
          int head = n >> 7, h = n & 127;  // n = head*HD + h
          __bf16 bv = (__bf16)v;
          if (mode == 1)
            ((__bf16*)dst)[(((size_t)(b * NH + head)) * SS + s) * HD + h] = bv;
          else
            ((__bf16*)dst)[(((size_t)(b * NH + head)) * HD + h) * SS + s] = bv;
        }
      }
}

// ---------------------------------------------------------------------------
// Prefill causal flash attention.
//   One wave per (b, head, 16-row query tile). 8 waves / block, independent.
//   Q,K: [b,head][S][HD] bf16 ;  V: [b,head][HD][S] bf16 (transposed)
//   O:   [b*S + s][DM] bf16 (row-major, ready to be GEMM-A of out-projection)
// ---------------------------------------------------------------------------
__global__ __launch_bounds__(256) void k_attn_prefill(
    const __bf16* __restrict__ Qb, const __bf16* __restrict__ Kb,
    const __bf16* __restrict__ Vt, __bf16* __restrict__ Ob) {
  __shared__ __bf16 plds[8][16 * 32];  // per-wave P tile (16 rows x 32 keys)

  int lane = threadIdx.x & 31;
  int wave = threadIdx.x >> 5;
  int li = lane & 15, hi = lane >> 4;
  int task = blockIdx.x * 8 + wave;  // 4096 tasks total
  int b = task >> 11;
  int rem = task & 2047;
  int head = rem >> 7;
  int qt = rem & 127;
  int m0 = qt * 16;
  const float scale = 0.08838834764831845f;  // 1/sqrt(128)

  const __bf16* Qh = Qb + ((size_t)(b * NH + head)) * SS * HD;
  const __bf16* Kh = Kb + ((size_t)(b * NH + head)) * SS * HD;
  const __bf16* Vh = Vt + ((size_t)(b * NH + head)) * HD * SS;

  // Q fragments: 16 rows x 128 dims as four 16x32 A-fragments
  v16bf qf[4];
  {
    const __bf16* qrow = Qh + (size_t)(m0 + li) * HD;
#pragma unroll
    for (int kc = 0; kc < 4; kc++) {
      const __bf16* p = qrow + kc * 32 + hi * 8;
      qf[kc] = cat16(*(const v8bf*)p, *(const v8bf*)(p + 16));
    }
  }

  v8f acc[8];  // 16 rows x 128 head-dims of output
#pragma unroll
  for (int i = 0; i < 8; i++)
#pragma unroll
    for (int r = 0; r < 8; r++) acc[i][r] = 0.0f;
  float mrow[8], lrow[8];
#pragma unroll
  for (int r = 0; r < 8; r++) { mrow[r] = -1e30f; lrow[r] = 0.0f; }

  int kend = m0 + 16;  // causal: keys < kend
  for (int kt = 0; kt < kend; kt += 32) {
    v8f s0, s1;
#pragma unroll
    for (int r = 0; r < 8; r++) { s0[r] = 0.0f; s1[r] = 0.0f; }
    bool do2 = (kt + 16) < kend;

    // scores = Q * K^T : B-fragment column n is key row (contiguous in h)
#pragma unroll
    for (int kc = 0; kc < 4; kc++) {
      const __bf16* p0 = Kh + (size_t)(kt + li) * HD + kc * 32 + hi * 8;
      v16bf b0 = cat16(*(const v8bf*)p0, *(const v8bf*)(p0 + 16));
      s0 = wmma_bf16(qf[kc], b0, s0);
    }
    if (do2) {
#pragma unroll
      for (int kc = 0; kc < 4; kc++) {
        const __bf16* p1 = Kh + (size_t)(kt + 16 + li) * HD + kc * 32 + hi * 8;
        v16bf b1 = cat16(*(const v8bf*)p1, *(const v8bf*)(p1 + 16));
        s1 = wmma_bf16(qf[kc], b1, s1);
      }
    }

    // online softmax per row; row r of component lives in lanes (hi half) 0..15
#pragma unroll
    for (int r = 0; r < 8; r++) {
      int row = m0 + r + hi * 8;
      int c0 = kt + li;
      float v0 = (c0 <= row) ? s0[r] * scale : -1e30f;
      float v1 = (do2 && (c0 + 16) <= row) ? s1[r] * scale : -1e30f;
      float mx = fmaxf(v0, v1);
#pragma unroll
      for (int off = 1; off < 16; off <<= 1) mx = fmaxf(mx, __shfl_xor(mx, off, 32));
      float mnew = fmaxf(mrow[r], mx);
      float alpha = __expf(mrow[r] - mnew);
      mrow[r] = mnew;
      float p0 = __expf(v0 - mnew);
      float p1 = __expf(v1 - mnew);
      float rs = p0 + p1;
#pragma unroll
      for (int off = 1; off < 16; off <<= 1) rs += __shfl_xor(rs, off, 32);
      lrow[r] = lrow[r] * alpha + rs;
#pragma unroll
      for (int ht = 0; ht < 8; ht++) acc[ht][r] *= alpha;
      int rl = r + hi * 8;
      plds[wave][rl * 32 + li] = (__bf16)p0;
      plds[wave][rl * 32 + li + 16] = (__bf16)p1;
    }

    // relayout P (C-layout -> A-fragment) via per-wave LDS slice
    asm volatile("s_wait_dscnt 0" ::: "memory");
    const __bf16* pl = &plds[wave][li * 32 + hi * 8];
    v16bf pf = cat16(*(const v8bf*)pl, *(const v8bf*)(pl + 16));

    // acc += P(16x32) * V(32x128) ; V transposed so B-frag columns are contiguous
#pragma unroll
    for (int ht = 0; ht < 8; ht++) {
      const __bf16* pv = Vh + (size_t)(ht * 16 + li) * SS + kt + hi * 8;
      v16bf vf = cat16(*(const v8bf*)pv, *(const v8bf*)(pv + 16));
      acc[ht] = wmma_bf16(pf, vf, acc[ht]);
    }
  }

  // normalize and write O row-major [b*S+s][head*HD+h]
#pragma unroll
  for (int r = 0; r < 8; r++) {
    float inv = 1.0f / lrow[r];
    int row = m0 + r + hi * 8;
#pragma unroll
    for (int ht = 0; ht < 8; ht++) {
      int h = head * HD + ht * 16 + li;
      Ob[((size_t)(b * SS + row)) * DM + h] = (__bf16)(acc[ht][r] * inv);
    }
  }
}

// ---------------------------------------------------------------------------
// Decode path (one token) in f32 — negligible cost, keep full precision.
// ---------------------------------------------------------------------------
__global__ void k_dec_proj(const float* __restrict__ xn,
                           const float* __restrict__ wq, const float* __restrict__ wk,
                           const float* __restrict__ wv,
                           float* __restrict__ q1, float* __restrict__ k1,
                           float* __restrict__ v1) {
  int idx = blockIdx.x * 256 + threadIdx.x;
  if (idx >= 3 * BB * DM) return;
  int mat = idx / (BB * DM);
  int rem = idx % (BB * DM);
  int b = rem >> 11, n = rem & 2047;
  const float* W = (mat == 0) ? wq : (mat == 1) ? wk : wv;
  const float* xr = xn + b * DM;
  float s = 0.0f;
  for (int d = 0; d < DM; d++) s += xr[d] * W[(size_t)d * DM + n];
  float* dst = (mat == 0) ? q1 : (mat == 1) ? k1 : v1;
  dst[b * DM + n] = s;
}

__global__ __launch_bounds__(256) void k_dec_attn(
    const float* __restrict__ q1, const float* __restrict__ k1,
    const float* __restrict__ v1, const __bf16* __restrict__ Kb,
    const __bf16* __restrict__ Vt, float* __restrict__ sc,
    float* __restrict__ qkv1) {
  int bh = blockIdx.x;  // b*16 + head
  int b = bh >> 4, head = bh & 15;
  __shared__ float q1s[HD];
  __shared__ float red[256];
  int tid = threadIdx.x;
  const float scale = 0.08838834764831845f;
  if (tid < HD) q1s[tid] = q1[b * DM + head * HD + tid];
  __syncthreads();

  const __bf16* Kh = Kb + (size_t)bh * SS * HD;
  float* scr = sc + (size_t)bh * (SS + 1);
  float lmax = -1e30f;
  for (int t = tid; t <= SS; t += 256) {
    float s = 0.0f;
    if (t < SS) {
      const __bf16* kr = Kh + (size_t)t * HD;
      for (int h = 0; h < HD; h++) s += q1s[h] * (float)kr[h];
    } else {
      const float* kr = k1 + b * DM + head * HD;
      for (int h = 0; h < HD; h++) s += q1s[h] * kr[h];
    }
    s *= scale;
    scr[t] = s;
    lmax = fmaxf(lmax, s);
  }
  red[tid] = lmax; __syncthreads();
  for (int off = 128; off > 0; off >>= 1) {
    if (tid < off) red[tid] = fmaxf(red[tid], red[tid + off]);
    __syncthreads();
  }
  float m = red[0]; __syncthreads();
  float lsum = 0.0f;
  for (int t = tid; t <= SS; t += 256) lsum += __expf(scr[t] - m);
  red[tid] = lsum; __syncthreads();
  for (int off = 128; off > 0; off >>= 1) {
    if (tid < off) red[tid] += red[tid + off];
    __syncthreads();
  }
  float invL = 1.0f / red[0];
  if (tid < HD) {
    int h = tid;
    const __bf16* Vhp = Vt + ((size_t)bh * HD + h) * SS;
    float a = 0.0f;
    for (int t = 0; t < SS; t++) a += __expf(scr[t] - m) * (float)Vhp[t];
    a += __expf(scr[SS] - m) * v1[b * DM + head * HD + h];
    qkv1[b * DM + head * HD + h] = a * invL;
  }
}

__global__ void k_dec_out(const float* __restrict__ qkv1,
                          const float* __restrict__ wo, float* __restrict__ dout) {
  int idx = blockIdx.x * 256 + threadIdx.x;
  if (idx >= BB * DM) return;
  int b = idx >> 11, d = idx & 2047;
  const float* a = qkv1 + b * DM;
  const float* w = wo + (size_t)d * DM;
  float s = 0.0f;
  for (int k2 = 0; k2 < DM; k2++) s += a[k2] * w[k2];
  dout[idx] = s;
}

// ---------------------------------------------------------------------------
// Workspace layout (bytes)
// ---------------------------------------------------------------------------
#define OFF_XB   ((size_t)0)                 // x bf16   [4096,2048]   16 MB
#define OFF_WQT  ((size_t)16777216)          // wq^T bf16 [2048,2048]   8 MB
#define OFF_WKT  ((size_t)25165824)
#define OFF_WVT  ((size_t)33554432)
#define OFF_WO   ((size_t)41943040)          // wo bf16 natural (== Bt for out-proj)
#define OFF_Q    ((size_t)50331648)          // Q bf16 [b,h][S][HD]    16 MB
#define OFF_K    ((size_t)67108864)
#define OFF_VT   ((size_t)83886080)          // V bf16 [b,h][HD][S]
#define OFF_O    ((size_t)100663296)         // attn out bf16 [4096][2048]
#define OFF_Q1   ((size_t)117440512)
#define OFF_K1   ((size_t)117456896)
#define OFF_V1   ((size_t)117473280)
#define OFF_QKV1 ((size_t)117489664)
#define OFF_SC   ((size_t)117506048)         // 32*(S+1)*4

extern "C" void kernel_launch(void* const* d_in, const int* in_sizes, int n_in,
                              void* d_out, int out_size, void* d_ws, size_t ws_size,
                              hipStream_t stream) {
  const float* x    = (const float*)d_in[0];
  const float* xnew = (const float*)d_in[1];
  const float* wq   = (const float*)d_in[2];
  const float* wk   = (const float*)d_in[3];
  const float* wv   = (const float*)d_in[4];
  const float* wo   = (const float*)d_in[5];
  float* out = (float*)d_out;

  char* ws = (char*)d_ws;
  __bf16* xb  = (__bf16*)(ws + OFF_XB);
  __bf16* wqT = (__bf16*)(ws + OFF_WQT);
  __bf16* wkT = (__bf16*)(ws + OFF_WKT);
  __bf16* wvT = (__bf16*)(ws + OFF_WVT);
  __bf16* wob = (__bf16*)(ws + OFF_WO);
  __bf16* Qb  = (__bf16*)(ws + OFF_Q);
  __bf16* Kb  = (__bf16*)(ws + OFF_K);
  __bf16* Vt  = (__bf16*)(ws + OFF_VT);
  __bf16* Ob  = (__bf16*)(ws + OFF_O);
  float* q1   = (float*)(ws + OFF_Q1);
  float* k1   = (float*)(ws + OFF_K1);
  float* v1   = (float*)(ws + OFF_V1);
  float* qkv1 = (float*)(ws + OFF_QKV1);
  float* sc   = (float*)(ws + OFF_SC);

  // 1) precision conversion / relayout
  int nx = MTOT * DM;           // 8388608
  k_cvt_bf16<<<(nx + 255) / 256, 256, 0, stream>>>(x, xb, nx);
  dim3 tgrid(DM / 32, DM / 32);
  k_transpose_bf16<<<tgrid, 256, 0, stream>>>(wq, wqT, DM);
  k_transpose_bf16<<<tgrid, 256, 0, stream>>>(wk, wkT, DM);
  k_transpose_bf16<<<tgrid, 256, 0, stream>>>(wv, wvT, DM);
  int nw = DM * DM;
  k_cvt_bf16<<<(nw + 255) / 256, 256, 0, stream>>>(wo, wob, nw);

  // 2) Q/K/V projections (bf16 WMMA GEMMs)
  dim3 ggrid(DM / 256, MTOT / 64);
  k_gemm_bf16<<<ggrid, 256, 0, stream>>>(xb, wqT, (void*)Qb, MTOT, DM, DM, 1);
  k_gemm_bf16<<<ggrid, 256, 0, stream>>>(xb, wkT, (void*)Kb, MTOT, DM, DM, 1);
  k_gemm_bf16<<<ggrid, 256, 0, stream>>>(xb, wvT, (void*)Vt, MTOT, DM, DM, 2);

  // 3) causal flash attention (WMMA)
  k_attn_prefill<<<(BB * NH * (SS / 16)) / 8, 256, 0, stream>>>(Qb, Kb, Vt, Ob);

  // 4) output projection -> prefill_out (f32)
  k_gemm_bf16<<<ggrid, 256, 0, stream>>>(Ob, wob, (void*)out, MTOT, DM, DM, 0);

  // 5) decode path
  k_dec_proj<<<(3 * BB * DM + 255) / 256, 256, 0, stream>>>(xnew, wq, wk, wv, q1, k1, v1);
  k_dec_attn<<<BB * NH, 256, 0, stream>>>(q1, k1, v1, Kb, Vt, sc, qkv1);
  k_dec_out<<<(BB * DM + 255) / 256, 256, 0, stream>>>(qkv1, wo, out + PRE_OUT);
}
